// PtrNet_48833778155990
// MI455X (gfx1250) — matrix operand order repaired and checked by reading, not compiled, not run on hardware
//
#include <hip/hip_runtime.h>
#include <math.h>

// Problem dims (fixed by the reference)
#define B_  512
#define S_  128
#define H_  256
#define G3_ 768   // 3*H

typedef __attribute__((ext_vector_type(16))) __bf16 v16bf;
typedef __attribute__((ext_vector_type(8)))  __bf16 v8bf;
typedef __attribute__((ext_vector_type(8)))  float  v8f;

// ---------------------------------------------------------------------------
// CDNA5 async global->LDS copy (ISA 15.18.3 opcode 98) + ASYNCcnt wait.
// VDST VGPR carries the LDS byte address; for a generic pointer to __shared__
// data the low 32 bits ARE the LDS offset.
__device__ __forceinline__ void async_copy_b128(void* lds_dst, const void* gsrc) {
  unsigned lds_addr = (unsigned)(size_t)lds_dst;
  asm volatile("global_load_async_to_lds_b128 %0, %1, off"
               :: "v"(lds_addr), "v"(gsrc) : "memory");
}
__device__ __forceinline__ void async_wait0() {
  asm volatile("s_wait_asynccnt 0x0" ::: "memory");
}

// ---------------------------------------------------------------------------
// WMMA fragment loaders (wave32, 16x16x32 bf16 -> f32)
//
// A matrix 16x32 bf16: lane L holds row M=L%16; element i
// -> K = k0 + (L>=16 ? 16 : 0) + i (contiguous, matches B)
// => two contiguous 16B loads per lane from row-major [row][K] storage.
__device__ __forceinline__ v16bf load_a_frag_lds(const __bf16* __restrict__ As,
                                                 int k0) {
  const int lane = threadIdx.x & 31;
  const __bf16* p = As + (lane & 15) * H_ + (k0 + ((lane >> 4) << 4));
  v8bf lo = *(const v8bf*)(p);
  v8bf hi = *(const v8bf*)(p + 8);
  v16bf f;
#pragma unroll
  for (int i = 0; i < 8; ++i) { f[i] = lo[i]; f[i + 8] = hi[i]; }
  return f;
}

// B matrix 32x16 bf16 (B = W^T with W row-major [j][k]): lane L covers column
// N = j0 + L%16, K = k0 + (L>=16 ? 16 : 0) + i for element i (contiguous)
// => one contiguous 32B load per lane (pair of global_load_b128).
__device__ __forceinline__ v16bf load_b_frag(const __bf16* __restrict__ W,
                                             int ldw, int j0, int k0) {
  const int lane = threadIdx.x & 31;
  const __bf16* p = W + (size_t)(j0 + (lane & 15)) * ldw
                      + (k0 + ((lane >> 4) << 4));
  return *(const v16bf*)(p);
}

__device__ __forceinline__ float sigmf(float x) {
  return 1.0f / (1.0f + __expf(-x));
}

// ---------------------------------------------------------------------------
// Setup kernels
__global__ void k_f32_to_bf16(const float* __restrict__ src,
                              __bf16* __restrict__ dst, int n) {
  int i = blockIdx.x * blockDim.x + threadIdx.x;
  if (i < n) dst[i] = (__bf16)src[i];
}

__global__ void k_zero_u32(unsigned* __restrict__ p, int n) {
  int i = blockIdx.x * blockDim.x + threadIdx.x;
  if (i < n) p[i] = 0u;
}

__global__ void k_init(float* __restrict__ mask, int* __restrict__ ptrc,
                       float* __restrict__ logp, int* __restrict__ ptrs) {
  int b = blockIdx.x * blockDim.x + threadIdx.x;
  if (b < B_) {
    ptrc[b] = 0;
    logp[b] = 0.0f;
    ptrs[b * S_] = 0;
    mask[b * S_] = -INFINITY;
    for (int s = 1; s < S_; ++s) mask[b * S_ + s] = 0.0f;
  }
}

// ---------------------------------------------------------------------------
// One GRU step for all 512 batch rows. All 8 waves of a block share one
// 16-row batch tile, so the block async-stages the 16x256 bf16 A tile (8KB,
// contiguous) into LDS once; each wave then owns a 16(b) x 16(h) output tile
// with 3 f32 accumulators (r/z/n slices of gh = h @ Whh^T), walking K=256 in
// 8 WMMA steps (24 v_wmma per wave). Double-buffered h -> no races.
// grid = 64 blocks x 256 threads; bt = blockIdx>>1, jt = (blockIdx&1)*8+waveInBlk.
__global__ __launch_bounds__(256)
void k_gru_step(const float* __restrict__ x_base,      // input [B][S][2]
                int t,                                  // timestep (encoder x)
                const int* __restrict__ ptr,            // decoder gather (or null)
                const __bf16* __restrict__ h_bf_in,     // [B][H]
                const float*  __restrict__ h_f32_in,    // [B][H]
                __bf16* __restrict__ h_bf_out,
                float*  __restrict__ h_f32_out,
                const __bf16* __restrict__ Whh_bf,      // [3H][H] row-major
                const float* __restrict__ Wih,          // [3H][2]
                const float* __restrict__ bih,          // [3H]
                const float* __restrict__ bhh,          // [3H]
                __bf16* __restrict__ enc_bf)            // [B][S][H] or null
{
  __shared__ __attribute__((aligned(32))) __bf16 As[16 * H_];   // 8KB A tile

  const int tid = threadIdx.x;
  const int bt = blockIdx.x >> 1;                       // 0..31 batch tile
  const int jt = ((blockIdx.x & 1) << 3) + (tid >> 5);  // 0..15 hidden tile
  const int lane = tid & 31;

  // ---- async stage of the shared A tile: 512 x 16B chunks ----
  {
    const __bf16* Abase = h_bf_in + (size_t)bt * 16 * H_;
    async_copy_b128(&As[(size_t)tid * 8],         Abase + (size_t)tid * 8);
    async_copy_b128(&As[(size_t)(tid + 256) * 8], Abase + (size_t)(tid + 256) * 8);
  }
  async_wait0();
  __syncthreads();

  const int j0r = jt * 16;
  const int j0z = jt * 16 + H_;
  const int j0n = jt * 16 + 2 * H_;

  v8f ar = {}, az = {}, an = {};
#pragma unroll
  for (int kt = 0; kt < 8; ++kt) {
    v16bf a  = load_a_frag_lds(As, kt * 32);
    v16bf br = load_b_frag(Whh_bf, H_, j0r, kt * 32);
    v16bf bz = load_b_frag(Whh_bf, H_, j0z, kt * 32);
    v16bf bn = load_b_frag(Whh_bf, H_, j0n, kt * 32);
    if (kt < 7)
      __builtin_prefetch((const void*)(Whh_bf + (size_t)j0r * H_ + (kt + 1) * 32), 0, 1);
    ar = __builtin_amdgcn_wmma_f32_16x16x32_bf16(false, a, false, br, (short)0, ar, false, false);
    az = __builtin_amdgcn_wmma_f32_16x16x32_bf16(false, a, false, bz, (short)0, az, false, false);
    an = __builtin_amdgcn_wmma_f32_16x16x32_bf16(false, a, false, bn, (short)0, an, false, false);
  }

  // Gate epilogue. C-fragment layout: element v at lane L holds
  // M = v + 8*(L>=16), N = L%16.
  const int nl = lane & 15;
  const int jcol = jt * 16 + nl;
  const int jr = j0r + nl, jz = j0z + nl, jn = j0n + nl;
  const float wr0 = Wih[jr * 2], wr1 = Wih[jr * 2 + 1];
  const float wz0 = Wih[jz * 2], wz1 = Wih[jz * 2 + 1];
  const float wn0 = Wih[jn * 2], wn1 = Wih[jn * 2 + 1];
  const float bir = bih[jr], biz = bih[jz], bin = bih[jn];
  const float bhr = bhh[jr], bhz = bhh[jz], bhn = bhh[jn];

#pragma unroll
  for (int v = 0; v < 8; ++v) {
    const int b = bt * 16 + v + ((lane >> 4) << 3);
    const int s = ptr ? ptr[b] : t;
    const float x0 = x_base[((size_t)b * S_ + s) * 2 + 0];
    const float x1 = x_base[((size_t)b * S_ + s) * 2 + 1];

    const float r  = sigmf(x0 * wr0 + x1 * wr1 + bir + ar[v] + bhr);
    const float z  = sigmf(x0 * wz0 + x1 * wz1 + biz + az[v] + bhz);
    const float nn = tanhf(x0 * wn0 + x1 * wn1 + bin + r * (an[v] + bhn));

    const float ho = h_f32_in[(size_t)b * H_ + jcol];
    const float hn = (1.0f - z) * nn + z * ho;

    h_f32_out[(size_t)b * H_ + jcol] = hn;
    h_bf_out[(size_t)b * H_ + jcol]  = (__bf16)hn;
    if (enc_bf) enc_bf[((size_t)b * S_ + t) * H_ + jcol] = (__bf16)hn;
  }
}

// ---------------------------------------------------------------------------
// Generic C[m][j] = A[m][:]@W[j][:]^T GEMM (K=256, N tiles of 16). Block stages
// its shared 16x256 A tile into LDS via async copy; B fragments stream from
// global (weights stay L2/WGP$-resident). Used for w1_e = enc_out @ W1^T
// (M=65536, 8192 blocks) and q = hn @ W2^T (M=512, 64 blocks).
__global__ __launch_bounds__(256)
void k_gemm_wt(const __bf16* __restrict__ A,   // [M][256] bf16 row-major
               const __bf16* __restrict__ W,   // [256][256] bf16 row-major
               float* __restrict__ C)          // [M][256] f32
{
  __shared__ __attribute__((aligned(32))) __bf16 As[16 * H_];   // 8KB A tile

  const int tid = threadIdx.x;
  const int mt = blockIdx.x >> 1;
  const int jt = ((blockIdx.x & 1) << 3) + (tid >> 5);
  const int lane = tid & 31;

  {
    const __bf16* Abase = A + (size_t)mt * 16 * H_;
    async_copy_b128(&As[(size_t)tid * 8],         Abase + (size_t)tid * 8);
    async_copy_b128(&As[(size_t)(tid + 256) * 8], Abase + (size_t)(tid + 256) * 8);
  }
  async_wait0();
  __syncthreads();

  v8f c = {};
#pragma unroll
  for (int kt = 0; kt < 8; ++kt) {
    v16bf a = load_a_frag_lds(As, kt * 32);
    v16bf b = load_b_frag(W, H_, jt * 16, kt * 32);
    if (kt < 7)
      __builtin_prefetch((const void*)(W + (size_t)(jt * 16) * H_ + (kt + 1) * 32), 0, 1);
    c = __builtin_amdgcn_wmma_f32_16x16x32_bf16(false, a, false, b, (short)0, c, false, false);
  }

  const int n = jt * 16 + (lane & 15);
#pragma unroll
  for (int v = 0; v < 8; ++v) {
    const int m = mt * 16 + v + ((lane >> 4) << 3);
    C[(size_t)m * H_ + n] = c[v];
  }
}

// ---------------------------------------------------------------------------
// Attention + softmax-argmax + pointer/mask/logp update for decoder step t.
// One block per batch row b; q-row and V async-staged to LDS; 8 waves each
// cover 16 of the 128 sequence positions.
// scores[s] = V . tanh(w1e[b,s,:] + q[b,:]) + mask[b,s]
// log(max softmax) = -log(sum exp(s - smax)); argmax(softmax) = argmax(scores).
__global__ __launch_bounds__(256)
void k_attn(const float* __restrict__ w1e,   // [B*S][256]
            const float* __restrict__ q,     // [B][256]
            const float* __restrict__ Vv,    // [256]
            float* __restrict__ mask,        // [B][S]
            int* __restrict__ ptrc,          // [B]
            float* __restrict__ logp,        // [B] (lives in d_out)
            int* __restrict__ ptrs_out,      // [B][S] (lives in d_out)
            int t)
{
  __shared__ __attribute__((aligned(16))) float qs[H_];
  __shared__ __attribute__((aligned(16))) float vs[H_];
  __shared__ float sc[S_];

  const int b = blockIdx.x;
  const int tid = threadIdx.x;

  // async stage q[b][:] (64 x 16B) and V (64 x 16B)
  if (tid < 128) {
    const float* g = (tid < 64) ? (q + (size_t)b * H_ + tid * 4)
                                : (Vv + (size_t)(tid - 64) * 4);
    float* l = (tid < 64) ? &qs[tid * 4] : &vs[(tid - 64) * 4];
    async_copy_b128(l, g);
  }
  async_wait0();
  __syncthreads();

  const int wv = tid >> 5, lane = tid & 31;
  for (int si = 0; si < 16; ++si) {
    const int s = wv * 16 + si;
    const float* row = w1e + ((size_t)b * S_ + s) * H_;
    float acc = 0.0f;
#pragma unroll
    for (int k = 0; k < 8; ++k) {
      const int h = lane + 32 * k;            // coalesced across the wave
      acc += vs[h] * tanhf(row[h] + qs[h]);
    }
#pragma unroll
    for (int off = 16; off > 0; off >>= 1) acc += __shfl_down(acc, off, 32);
    if (lane == 0) sc[s] = acc + mask[(size_t)b * S_ + s];
  }
  __syncthreads();

  if (wv == 0) {
    float best = -INFINITY; int bi = 0x7fffffff;
#pragma unroll
    for (int k = 0; k < 4; ++k) {
      const int s = lane + 32 * k;
      const float v = sc[s];
      if (v > best || (v == best && s < bi)) { best = v; bi = s; }
    }
#pragma unroll
    for (int off = 16; off > 0; off >>= 1) {
      const float ob = __shfl_down(best, off, 32);
      const int   oi = __shfl_down(bi,   off, 32);
      if (ob > best || (ob == best && oi < bi)) { best = ob; bi = oi; }
    }
    best = __shfl(best, 0, 32);
    bi   = __shfl(bi,   0, 32);

    float se = 0.0f;
#pragma unroll
    for (int k = 0; k < 4; ++k) se += __expf(sc[lane + 32 * k] - best);
#pragma unroll
    for (int off = 16; off > 0; off >>= 1) se += __shfl_down(se, off, 32);

    if (lane == 0) {
      logp[b] += -__logf(se);
      ptrc[b] = bi;
      mask[(size_t)b * S_ + bi] = -INFINITY;
      ptrs_out[(size_t)b * S_ + (t + 1)] = bi;
    }
  }
}

// ---------------------------------------------------------------------------
extern "C" void kernel_launch(void* const* d_in, const int* in_sizes, int n_in,
                              void* d_out, int out_size, void* d_ws, size_t ws_size,
                              hipStream_t stream) {
  (void)in_sizes; (void)n_in; (void)out_size; (void)ws_size;

  const float* input   = (const float*)d_in[0];   // [512][128][2]
  const float* encWih  = (const float*)d_in[1];   // [768][2]
  const float* encWhh  = (const float*)d_in[2];   // [768][256]
  const float* encBih  = (const float*)d_in[3];
  const float* encBhh  = (const float*)d_in[4];
  const float* decWih  = (const float*)d_in[5];
  const float* decWhh  = (const float*)d_in[6];
  const float* decBih  = (const float*)d_in[7];
  const float* decBhh  = (const float*)d_in[8];
  const float* W1      = (const float*)d_in[9];   // [256][256]
  const float* W2      = (const float*)d_in[10];  // [256][256]
  const float* Vv      = (const float*)d_in[11];  // [256]

  float* logp    = (float*)d_out;                 // [512]
  int*   ptrsOut = (int*)((float*)d_out + B_);    // [512][128]

  // ---- workspace carve-out (256B aligned) ----
  char* w = (char*)d_ws;
  size_t off = 0;
  auto alloc = [&](size_t bytes) -> void* {
    off = (off + 255) & ~(size_t)255;
    void* p = w + off;
    off += bytes;
    return p;
  };
  __bf16* whhEb = (__bf16*)alloc((size_t)G3_ * H_ * 2);
  __bf16* whhDb = (__bf16*)alloc((size_t)G3_ * H_ * 2);
  __bf16* w1b   = (__bf16*)alloc((size_t)H_ * H_ * 2);
  __bf16* w2b   = (__bf16*)alloc((size_t)H_ * H_ * 2);
  __bf16* hbf0  = (__bf16*)alloc((size_t)B_ * H_ * 2);
  __bf16* hbf1  = (__bf16*)alloc((size_t)B_ * H_ * 2);
  float*  hf0   = (float*) alloc((size_t)B_ * H_ * 4);
  float*  hf1   = (float*) alloc((size_t)B_ * H_ * 4);
  __bf16* encbf = (__bf16*)alloc((size_t)B_ * S_ * H_ * 2);
  float*  w1e   = (float*) alloc((size_t)B_ * S_ * H_ * 4);
  float*  qbuf  = (float*) alloc((size_t)B_ * H_ * 4);
  float*  mask  = (float*) alloc((size_t)B_ * S_ * 4);
  int*    ptrc  = (int*)   alloc((size_t)B_ * 4);

  __bf16* hbf[2] = { hbf0, hbf1 };
  float*  hf[2]  = { hf0,  hf1  };

  // ---- setup: bf16 weight copies, zero h0, init mask/ptr/logp/pointers ----
  const int n3h = G3_ * H_, nhh = H_ * H_;
  k_f32_to_bf16<<<(n3h + 255) / 256, 256, 0, stream>>>(encWhh, whhEb, n3h);
  k_f32_to_bf16<<<(n3h + 255) / 256, 256, 0, stream>>>(decWhh, whhDb, n3h);
  k_f32_to_bf16<<<(nhh + 255) / 256, 256, 0, stream>>>(W1, w1b, nhh);
  k_f32_to_bf16<<<(nhh + 255) / 256, 256, 0, stream>>>(W2, w2b, nhh);
  k_zero_u32<<<(B_ * H_ / 2 + 255) / 256, 256, 0, stream>>>((unsigned*)hbf0, B_ * H_ / 2);
  k_zero_u32<<<(B_ * H_ + 255) / 256, 256, 0, stream>>>((unsigned*)hf0, B_ * H_);
  k_init<<<2, 256, 0, stream>>>(mask, ptrc, logp, ptrsOut);

  // ---- encoder GRU: 128 stream-ordered steps, double-buffered h ----
  for (int t = 0; t < S_; ++t) {
    k_gru_step<<<64, 256, 0, stream>>>(input, t, nullptr,
                                       hbf[t & 1], hf[t & 1],
                                       hbf[1 - (t & 1)], hf[1 - (t & 1)],
                                       whhEb, encWih, encBih, encBhh, encbf);
  }
  // h_last ends in buffer 0 (128 steps -> even parity).

  // ---- w1_e = enc_out @ W1^T : M = 65536 rows -> 4096 mtiles x 2 blocks ----
  k_gemm_wt<<<8192, 256, 0, stream>>>(encbf, w1b, w1e);

  // ---- autoregressive pointer decode: 127 steps x (GRU, q-GEMM, attention) ----
  for (int t = 0; t < S_ - 1; ++t) {
    k_gru_step<<<64, 256, 0, stream>>>(input, t, ptrc,
                                       hbf[t & 1], hf[t & 1],
                                       hbf[1 - (t & 1)], hf[1 - (t & 1)],
                                       whhDb, decWih, decBih, decBhh, nullptr);
    k_gemm_wt<<<64, 256, 0, stream>>>(hbf[1 - (t & 1)], w2b, qbuf);
    k_attn<<<B_, 256, 0, stream>>>(w1e, qbuf, Vv, mask, ptrc, logp, ptrsOut, t);
  }
}
